// RowLSTM_15564961481140
// MI455X (gfx1250) — matrix operand order repaired and checked by reading, not compiled, not run on hardware
//
#include <hip/hip_runtime.h>
#include <hip/hip_bf16.h>

typedef _Float16 v16h __attribute__((ext_vector_type(16)));
typedef float    v8f  __attribute__((ext_vector_type(8)));

#define HID   256
#define GATES 1024        // 4*HID
#define KTOT  512         // C (256) + HID (256)
#define NTOT  1024        // B*W = 16*64
#define HROWS 64

// ---------------- workspace layout (bytes) ----------------
#define WCAT_OFF 0u                          // 1024*512 f16   = 1 MB
#define BIAS_OFF 1048576u                    // 1024 f32       = 4 KB
#define XT_OFF   1052672u                    // 64*1024*256 f16 = 32 MB
#define HT_OFF   34607104u                   // 2 * 1024*256 f16 (ping-pong) = 1 MB
#define CS_OFF   35655680u                   // 256*1024 f32   = 1 MB

__device__ __forceinline__ float fsigmoid(float x) {
    return 1.0f / (1.0f + __expf(-x));
}

// CDNA5 async global->LDS (per-lane 16B copy, tracked by ASYNCcnt)
__device__ __forceinline__ void async_ld_b128(unsigned ldsOff, const void* g) {
    asm volatile("global_load_async_to_lds_b128 %0, %1, off"
                 :: "v"(ldsOff), "v"(g) : "memory");
}
__device__ __forceinline__ void wait_async0() {
    asm volatile("s_wait_asynccnt 0" ::: "memory");
}

// ---------------------------------------------------------------------------
// prep: Wcat = f16([Wx | Wh]) row-major (1024 x 512); bias = bx+bh;
//       zero c-state and h ping-pong buffer 0.
// ---------------------------------------------------------------------------
__global__ __launch_bounds__(256) void prep_kernel(
    const float* __restrict__ Wx, const float* __restrict__ Wh,
    const float* __restrict__ bx, const float* __restrict__ bh,
    _Float16* __restrict__ Wcat, float* __restrict__ bias,
    _Float16* __restrict__ Ht0, float* __restrict__ Cs)
{
    int i = blockIdx.x * 256 + threadIdx.x;     // 0 .. 1024*512-1
    int g = i >> 9;
    int k = i & 511;
    float v = (k < 256) ? Wx[g * 256 + k] : Wh[g * 256 + (k - 256)];
    Wcat[i] = (_Float16)v;
    if (i < GATES) bias[i] = bx[i] + bh[i];
    if (i < HID * NTOT) {                        // 262144
        Cs[i]  = 0.0f;
        Ht0[i] = (_Float16)0.0f;
    }
}

// ---------------------------------------------------------------------------
// xt: Xt[h][(b*64+w)][c] = f16(x[b][c][h][w]).  One block per (b,h).
// LDS-tiled so global reads are coalesced over w and writes are contiguous.
// ---------------------------------------------------------------------------
__global__ __launch_bounds__(256) void xt_kernel(
    const float* __restrict__ x, _Float16* __restrict__ Xt)
{
    __shared__ __align__(16) _Float16 lds[64 * 264];   // 264-f16 padded rows
    int bid = blockIdx.x;            // 0..1023
    int b = bid >> 6;
    int h = bid & 63;
    int t = threadIdx.x;
    long xbase = (long)b * (256 * 4096) + h * 64;      // x[b][c][h][w] fp32
#pragma unroll 4
    for (int it = 0; it < 64; ++it) {
        int idx = it * 256 + t;      // 0..16383
        int c = idx >> 6;
        int w = idx & 63;
        lds[w * 264 + c] = (_Float16)x[xbase + (long)c * 4096 + w];
    }
    __syncthreads();
    // copy 64 contiguous f16 per thread: row w2, c-chunk of 64
    int w2 = t >> 2;
    int cc = (t & 3) * 64;
    const uint4* s = (const uint4*)(lds + w2 * 264 + cc);
    uint4* d = (uint4*)(Xt + (long)h * (NTOT * 256) + b * (64 * 256) + w2 * 256 + cc);
#pragma unroll
    for (int j = 0; j < 8; ++j) d[j] = s[j];
}

// ---------------------------------------------------------------------------
// step: one LSTM row.  gates(1024 x 1024) = Wcat(1024x512) * [Xt[h]; Hin](512x1024)
// Block tile: M = 4 gates x 32 hid = 128 rows, N = 128.  8 waves, each wave
// 16(M) x 128(N) via 8 accumulators of v_wmma_f32_16x16x32_f16, K-loop of 16.
// Double-buffered LDS tiles fed by GLOBAL_LOAD_ASYNC_TO_LDS_B128 so the fetch
// of tile k+1 overlaps WMMAs on tile k; one barrier per K-step.
// Fused LSTM elementwise; h -> Hout (f16, ping-pong) + d_out (fp32).
// ---------------------------------------------------------------------------
__global__ __launch_bounds__(256) void step_kernel(
    const _Float16* __restrict__ Wcat, const float* __restrict__ bias,
    const _Float16* __restrict__ Xt,
    const _Float16* __restrict__ Hin, _Float16* __restrict__ Hout,
    float* __restrict__ Cs, float* __restrict__ out, int h)
{
    // buf0: A@0 (8KB), B@8192 (8KB); buf1: A@16384, B@24576; gst reuses all.
    __shared__ __align__(16) unsigned char shraw[64 * 132 * 4];  // 33792 B
    float* gst = (float*)shraw;                        // 64 x 132 f32 (reused)

    int tid   = threadIdx.x;
    int nBase = blockIdx.x * 128;      // N tile
    int hidB  = blockIdx.y * 32;       // hid tile (per gate)

    int lane = tid & 31;
    int lh   = lane >> 4;              // lane half
    int ln   = lane & 15;
    int wv   = tid >> 5;               // wave 0..7
    int gt   = wv >> 1;                // gate 0..3 handled by this wave
    int mSub = wv & 1;                 // 16-row half of the 32-hid tile

    // C init = bias broadcast (C layout: VGPR i -> M = i + 8*lh, N = ln)
    v8f binit;
#pragma unroll
    for (int i = 0; i < 8; ++i)
        binit[i] = bias[gt * 256 + hidB + mSub * 16 + i + 8 * lh];
    v8f acc[8];
#pragma unroll
    for (int nt = 0; nt < 8; ++nt) acc[nt] = binit;

    const _Float16* Xrow = Xt + (long)h * (NTOT * 256);

    int srow = tid >> 1;               // staging row 0..127
    int shal = tid & 1;                // 16-f16 half of a 32-k row
    int sgt  = srow >> 5;
    int shl  = srow & 31;
    const _Float16* aRow = Wcat + (long)(sgt * 256 + hidB + shl) * KTOT;

    unsigned ldsBase = (unsigned)(uintptr_t)(void*)shraw;
    unsigned stgOff  = (unsigned)((srow * 32 + shal * 16) * 2);  // byte offset in tile

    auto stage = [&](int kk, int buf) {
        int k0 = kk * 32;
        unsigned dstA = ldsBase + (unsigned)(buf * 16384) + stgOff;
        async_ld_b128(dstA, (const void*)(aRow + k0 + shal * 16));
        const _Float16* bS = (k0 < 256)
            ? (Xrow + (long)(nBase + srow) * 256 + k0 + shal * 16)
            : (Hin  + (long)(nBase + srow) * 256 + (k0 - 256) + shal * 16);
        async_ld_b128(dstA + 8192u, (const void*)bS);
    };

    union Frag { uint4 u[2]; v16h v; };

    // prologue: fetch tile 0
    stage(0, 0);
    wait_async0();
    __syncthreads();

    for (int kk = 0; kk < 16; ++kk) {
        if (kk + 1 < 16) stage(kk + 1, (kk + 1) & 1);   // overlap next fetch

        const _Float16* ldsA = (const _Float16*)(shraw + (kk & 1) * 16384);
        const _Float16* ldsB = ldsA + 4096;

        // ---- A fragment: lane m holds K {8*lh..+7} U {+16..+23} ----
        int mrow = gt * 32 + mSub * 16 + ln;
        Frag fa;
        fa.u[0] = *(const uint4*)(ldsA + mrow * 32 + lh * 8);
        fa.u[1] = *(const uint4*)(ldsA + mrow * 32 + lh * 8 + 16);

#pragma unroll
        for (int nt = 0; nt < 8; ++nt) {
            // B fragment: lane n holds column n, K {8*lh..+7} U {+16..+23}
            Frag fb;
            int col = nt * 16 + ln;
            fb.u[0] = *(const uint4*)(ldsB + col * 32 + lh * 8);
            fb.u[1] = *(const uint4*)(ldsB + col * 32 + lh * 8 + 16);
            acc[nt] = __builtin_amdgcn_wmma_f32_16x16x32_f16(
                false, fa.v, false, fb.v, (short)0, acc[nt], false, false);
        }

        wait_async0();          // tile kk+1 resident in LDS
        __syncthreads();        // all waves done reading tile kk
    }

    // ---- fused LSTM elementwise, two 16-hid phases through LDS ----
    for (int p = 0; p < 2; ++p) {
        __syncthreads();
        if (mSub == p) {
#pragma unroll
            for (int nt = 0; nt < 8; ++nt)
#pragma unroll
                for (int i = 0; i < 8; ++i)
                    gst[(gt * 16 + i + 8 * lh) * 132 + nt * 16 + ln] = acc[nt][i];
        }
        __syncthreads();
#pragma unroll
        for (int j = 0; j < 8; ++j) {
            int e  = j * 256 + tid;        // 0..2047
            int hl = e >> 7;               // 0..15
            int nl = e & 127;
            float iv = fsigmoid(gst[(hl)      * 132 + nl]);
            float fv = fsigmoid(gst[(16 + hl) * 132 + nl]);
            float ov = fsigmoid(gst[(32 + hl) * 132 + nl]);
            float gv = tanhf  (gst[(48 + hl) * 132 + nl]);
            int hid = hidB + p * 16 + hl;
            int n   = nBase + nl;
            int ci  = hid * NTOT + n;
            float cn = fv * Cs[ci] + iv * gv;
            Cs[ci] = cn;
            float hn = ov * tanhf(cn);
            Hout[n * 256 + hid] = (_Float16)hn;
            int b = n >> 6, w = n & 63;
            out[(((long)b * 256 + hid) * 64 + h) * 64 + w] = hn;
        }
    }
}

// ---------------------------------------------------------------------------
extern "C" void kernel_launch(void* const* d_in, const int* in_sizes, int n_in,
                              void* d_out, int out_size, void* d_ws, size_t ws_size,
                              hipStream_t stream)
{
    const float* x  = (const float*)d_in[0];
    const float* Wx = (const float*)d_in[1];
    const float* bx = (const float*)d_in[2];
    const float* Wh = (const float*)d_in[3];
    const float* bh = (const float*)d_in[4];
    float* out = (float*)d_out;
    (void)in_sizes; (void)n_in; (void)out_size; (void)ws_size;

    char* ws = (char*)d_ws;
    _Float16* Wcat = (_Float16*)(ws + WCAT_OFF);
    float*    bias = (float*)   (ws + BIAS_OFF);
    _Float16* Xt   = (_Float16*)(ws + XT_OFF);
    _Float16* Htp  = (_Float16*)(ws + HT_OFF);     // ping-pong: 2 x 262144 f16
    float*    Cs   = (float*)   (ws + CS_OFF);

    prep_kernel<<<2048, 256, 0, stream>>>(Wx, Wh, bx, bh, Wcat, bias, Htp, Cs);
    xt_kernel<<<1024, 256, 0, stream>>>(x, Xt);

    for (int h = 0; h < HROWS; ++h) {
        _Float16* hin  = Htp + (size_t)(h & 1) * (NTOT * 256);
        _Float16* hout = Htp + (size_t)((h + 1) & 1) * (NTOT * 256);
        step_kernel<<<dim3(8, 8), 256, 0, stream>>>(
            Wcat, bias, Xt, hin, hout, Cs, out, h);
    }
}
